// DilatedConv_39084202394116
// MI455X (gfx1250) — matrix operand order repaired and checked by reading, not compiled, not run on hardware
//
#include <hip/hip_runtime.h>

// Problem constants (match reference)
#define B_      16
#define C_      256
#define NELEM   8192
#define NDIL    256
#define NSTEPS  7935          // NELEM - ((n_filt-1)*n_dil + 1)
#define TPAD    7936          // 496 * 16, padded t range for the c buffer
#define TTILES  496
#define LCHUNK  128           // scan chunk length (= 2^7 -> 7 squarings)
#define GCHUNKS 62            // 62 * 128 = 7936 >= NSTEPS
#define HSTRIDE 260           // LDS row stride for hT[b][c]: 260 % 64 == 4 -> conflict-free

typedef __attribute__((ext_vector_type(16))) __bf16 v16bf;
typedef __attribute__((ext_vector_type(8)))  float  v8f;
typedef __attribute__((ext_vector_type(2)))  float  v2f;
typedef __attribute__((ext_vector_type(4)))  unsigned int v4u;
typedef __attribute__((ext_vector_type(8)))  int    v8i;
typedef __attribute__((ext_vector_type(4)))  int    v4i;

static __device__ __forceinline__ v8f wmma_bf16(v16bf a, v16bf b, v8f c) {
    return __builtin_amdgcn_wmma_f32_16x16x32_bf16(false, a, false, b, (short)0, c, false, false);
}

static __device__ __forceinline__ v8f wmma_f32(v2f a, v2f b, v8f c) {
    // 8 args: (neg_a, A, neg_b, B, c_mod, C, reuse_a, reuse_b)
    return __builtin_amdgcn_wmma_f32_16x16x4_f32(false, a, false, b, (short)0, c, false, false);
}

// ---------------------------------------------------------------------------
// Prep: swizzle weight[:,:,f] into bf16 A-fragment order (16x32 bf16 A tile):
//   lane = half*16 + m ; element e: K = (e/8)*16 + half*8 + (e%8)
// ---------------------------------------------------------------------------
__global__ void prep_wfrag_kernel(const float* __restrict__ weight,
                                  __bf16* __restrict__ wf) {
    const int f  = blockIdx.x;   // filter tap 0..1
    const int ot = blockIdx.y;   // o-tile 0..15
    for (int i = threadIdx.x; i < 8 * 32 * 16; i += blockDim.x) {
        const int kc   = i >> 9;
        const int lane = (i >> 4) & 31;
        const int e    = i & 15;
        const int o    = ot * 16 + (lane & 15);
        const int k    = kc * 32 + ((e >> 3) << 4) + ((lane >> 4) << 3) + (e & 7);
        const float v  = weight[(size_t)(o * 256 + k) * 2 + f];
        wf[(size_t)f * 65536 + ((size_t)(ot * 8 + kc) * 32 + lane) * 16 + e] = (__bf16)v;
    }
}

// ---------------------------------------------------------------------------
// Extract W0 = weight[:,:,0] as a dense f32 [o][c] matrix.
// ---------------------------------------------------------------------------
__global__ void extract_w0_kernel(const float* __restrict__ weight,
                                  float* __restrict__ P) {
    const int i = blockIdx.x * 256 + threadIdx.x;   // 65536 elements
    P[i] = weight[(size_t)i * 2 + 0];
}

// ---------------------------------------------------------------------------
// D = A @ A for a 256x256 f32 matrix, via V_WMMA_F32_16X16X4_F32.
// ---------------------------------------------------------------------------
__global__ __launch_bounds__(32) void matsq_kernel(const float* __restrict__ A,
                                                   float* __restrict__ D) {
    const int lane = threadIdx.x;
    const int i0 = blockIdx.x * 16, j0 = blockIdx.y * 16;
    const int n = lane & 15, half = lane >> 4;
    v8f acc = {};
    for (int kc = 0; kc < 64; ++kc) {
        v2f a, b;
        #pragma unroll
        for (int v = 0; v < 2; ++v) {
            const int k = kc * 4 + v + 2 * half;
            a[v] = A[(size_t)(i0 + n) * 256 + k];
            b[v] = A[(size_t)k * 256 + j0 + n];
        }
        acc = wmma_f32(a, b, acc);
    }
    #pragma unroll
    for (int r = 0; r < 8; ++r)
        D[(size_t)(i0 + r + 8 * half) * 256 + j0 + n] = acc[r];
}

// ---------------------------------------------------------------------------
// Swizzle a dense f32 [o][c] matrix into f32 16x16x4 A-fragment order.
// ---------------------------------------------------------------------------
__global__ void prep_f32frag_kernel(const float* __restrict__ M,
                                    float* __restrict__ wf) {
    const int ot = blockIdx.x;
    for (int i = threadIdx.x; i < 4096; i += blockDim.x) {
        const int kc = i >> 6, lane = (i >> 1) & 31, v = i & 1;
        const int o = ot * 16 + (lane & 15);
        const int k = kc * 4 + v + 2 * (lane >> 4);
        wf[(size_t)(ot * 64 + kc) * 64 + lane * 2 + v] = M[(size_t)o * 256 + k];
    }
}

// ---------------------------------------------------------------------------
// Phase 1: c[t,b,o] = sum_c W1[o,c] * x[b,c,t+256]  (bf16 WMMA, f32 accum)
// Stored in the scan's C-fragment order: cbuf[t*4096 + ot*256 + lane*8 + r].
// ---------------------------------------------------------------------------
__global__ __launch_bounds__(512) void conv_c_kernel(const float* __restrict__ x,
                                                     const __bf16* __restrict__ w1f,
                                                     float* __restrict__ cbuf) {
    __shared__ float lds[16 * 16 * 16];   // [t_local][o_local][b]
    const int tid  = threadIdx.x;
    const int lane = tid & 31;
    const int b    = tid >> 5;            // wave id == batch index
    const int ot   = blockIdx.y;
    const int t0   = blockIdx.x * 16;

    v8f acc = {};
    #pragma unroll
    for (int kc = 0; kc < 8; ++kc) {
        v16bf a = *(const v16bf*)(w1f + ((size_t)(ot * 8 + kc) * 32 + lane) * 16);
        const float* xp = x + (size_t)(b * 256 + kc * 32 + ((lane >> 4) << 3)) * NELEM + NDIL + t0 + (lane & 15);
        v16bf bb;
        #pragma unroll
        for (int s = 0; s < 16; ++s)
            bb[s] = (__bf16)xp[(size_t)(((s >> 3) << 4) + (s & 7)) * NELEM];
        acc = wmma_bf16(a, bb, acc);
    }

    const int half = lane >> 4;
    const int n    = lane & 15;
    #pragma unroll
    for (int r = 0; r < 8; ++r)
        lds[(n * 16 + (r + half * 8)) * 16 + b] = acc[r];
    __syncthreads();

    for (int i = tid; i < 4096; i += 512) {
        const int t   = i >> 8;
        const int rem = i & 255;
        const int l2  = rem >> 3;
        const int r2  = rem & 7;
        const int m   = r2 + ((l2 >> 4) << 3);
        cbuf[(size_t)(t0 + t) * 4096 + ot * 256 + rem] =
            lds[(t * 16 + m) * 16 + (l2 & 15)];
    }
}

// ---------------------------------------------------------------------------
// Shared scan-step machinery (f32 WMMA 16x16x4).
// hT is double-buffered in LDS as [b][c] f32, row stride HSTRIDE
// (260 % 64 == 4 -> conflict-free). One barrier per step.
// ---------------------------------------------------------------------------
static __device__ __forceinline__ void load_afrags(const float* __restrict__ wf,
                                                   int ot, int lane, v2f af[64]) {
    #pragma unroll
    for (int kc = 0; kc < 64; ++kc)
        af[kc] = *(const v2f*)(wf + (size_t)(ot * 64 + kc) * 64 + lane * 2);
}

static __device__ __forceinline__ v8f step_matmul(const v2f af[64],
                                                  const float* __restrict__ hT,
                                                  int lane, v8f acc) {
    const int n = lane & 15, half = lane >> 4;
    #pragma unroll
    for (int kc = 0; kc < 64; ++kc) {
        v2f b = *(const v2f*)(hT + n * HSTRIDE + kc * 4 + 2 * half);
        acc = wmma_f32(af[kc], b, acc);
    }
    return acc;
}

static __device__ __forceinline__ void store_h(float* __restrict__ hT,
                                               int ot, int lane, v8f acc) {
    const int n = lane & 15, half = lane >> 4;
    float* p = hT + n * HSTRIDE + ot * 16 + 8 * half;   // 16B-aligned
    float4 lo, hi;
    lo.x = acc[0]; lo.y = acc[1]; lo.z = acc[2]; lo.w = acc[3];
    hi.x = acc[4]; hi.y = acc[5]; hi.z = acc[6]; hi.w = acc[7];
    *(float4*)(p)     = lo;
    *(float4*)(p + 4) = hi;
}

// ---------------------------------------------------------------------------
// Pass 1: zero-init (chunk 0: h0) local scans; emit chunk-end states E_g.
// ---------------------------------------------------------------------------
__global__ __launch_bounds__(512) void scan_pass1_kernel(const float* __restrict__ x,
                                                         const float* __restrict__ w0f,
                                                         const float* __restrict__ cbuf,
                                                         float* __restrict__ estates) {
    __shared__ float hA[16 * HSTRIDE];
    __shared__ float hB[16 * HSTRIDE];
    const int tid = threadIdx.x, lane = tid & 31, ot = tid >> 5;
    const int g = blockIdx.x;

    v2f af[64];
    load_afrags(w0f, ot, lane, af);

    for (int i = tid; i < 4096; i += 512)
        hA[(i >> 8) * HSTRIDE + (i & 255)] = (g == 0) ? x[(size_t)i * NELEM] : 0.0f;
    __syncthreads();

    float* cur = hA;
    float* nxt = hB;
    const int base = g * LCHUNK;
    for (int i = 0; i < LCHUNK; ++i) {
        const float* cp = cbuf + (size_t)(base + i) * 4096 + ot * 256 + lane * 8;
        v8f acc = *(const v8f*)cp;
        if (i + 1 < LCHUNK) __builtin_prefetch(cp + 4096, 0, 0);
        acc = step_matmul(af, cur, lane, acc);
        store_h(nxt, ot, lane, acc);
        __syncthreads();
        float* t = cur; cur = nxt; nxt = t;
    }
    for (int i = tid; i < 4096; i += 512)
        estates[(size_t)g * 4096 + i] = cur[(i >> 8) * HSTRIDE + (i & 255)];
}

// ---------------------------------------------------------------------------
// Pass 2: boundary recurrence s_{g+1} = W0^L s_g + E_g (single block).
// ---------------------------------------------------------------------------
__global__ __launch_bounds__(512) void scan_pass2_kernel(const float* __restrict__ w0Lf,
                                                         const float* __restrict__ estates,
                                                         float* __restrict__ sstates) {
    __shared__ float hA[16 * HSTRIDE];
    __shared__ float hB[16 * HSTRIDE];
    const int tid = threadIdx.x, lane = tid & 31, ot = tid >> 5;
    const int n = lane & 15, half = lane >> 4;

    v2f af[64];
    load_afrags(w0Lf, ot, lane, af);

    for (int i = tid; i < 4096; i += 512) {
        const float v = estates[i];                 // E_0 == s_1
        hA[(i >> 8) * HSTRIDE + (i & 255)] = v;
        sstates[4096 + i] = v;
    }
    __syncthreads();

    float* cur = hA;
    float* nxt = hB;
    for (int g = 1; g <= GCHUNKS - 2; ++g) {
        const size_t cbase = (size_t)g * 4096 + n * 256 + ot * 16 + 8 * half;
        v8f acc = *(const v8f*)(estates + cbase);   // E_g, [b][c] layout
        acc = step_matmul(af, cur, lane, acc);
        store_h(nxt, ot, lane, acc);
        *(v8f*)(sstates + (size_t)(g + 1) * 4096 + n * 256 + ot * 16 + 8 * half) = acc;
        __syncthreads();
        float* t = cur; cur = nxt; nxt = t;
    }
}

// ---------------------------------------------------------------------------
// Pass 3: definitive scans from s_g; write ys[t] in place over cbuf[t].
// ---------------------------------------------------------------------------
__global__ __launch_bounds__(512) void scan_pass3_kernel(const float* __restrict__ x,
                                                         const float* __restrict__ w0f,
                                                         const float* __restrict__ sstates,
                                                         float* __restrict__ cbuf) {
    __shared__ float hA[16 * HSTRIDE];
    __shared__ float hB[16 * HSTRIDE];
    const int tid = threadIdx.x, lane = tid & 31, ot = tid >> 5;
    const int g = blockIdx.x;

    v2f af[64];
    load_afrags(w0f, ot, lane, af);

    for (int i = tid; i < 4096; i += 512)
        hA[(i >> 8) * HSTRIDE + (i & 255)] =
            (g == 0) ? x[(size_t)i * NELEM] : sstates[(size_t)g * 4096 + i];
    __syncthreads();

    float* cur = hA;
    float* nxt = hB;
    const int base = g * LCHUNK;
    for (int i = 0; i < LCHUNK; ++i) {
        const int t = base + i;
        if (t >= NSTEPS) break;                     // uniform across block
        float* cp = cbuf + (size_t)t * 4096 + ot * 256 + lane * 8;
        v8f acc = *(const v8f*)cp;
        if (t + 1 < NSTEPS && i + 1 < LCHUNK) __builtin_prefetch(cp + 4096, 0, 0);
        acc = step_matmul(af, cur, lane, acc);
        store_h(nxt, ot, lane, acc);
        *(v8f*)cp = acc;                            // ys[t], fragment order
        __syncthreads();
        float* tp = cur; cur = nxt; nxt = tp;
    }
}

// ---------------------------------------------------------------------------
// Phase 3: ys (fragment order) -> out[b][o][1+t] via 32x32 LDS tiles.
// Tile load done by the Tensor Data Mover: 2D D# with tensor_dim0=4096,
// tile 32x32 f32, and pad_enable (interval=32 DWORDs, amount=1 DWORD) so the
// TDM itself writes the tile with a 33-float row stride (bank-conflict-free).
// ---------------------------------------------------------------------------
__global__ __launch_bounds__(256) void transpose_out_kernel(const float* __restrict__ ys,
                                                            float* __restrict__ out) {
    __shared__ float tile[32 * 33];
    const int t0 = blockIdx.x * 32;
    const int q0 = blockIdx.y * 32;
    const int tx = threadIdx.x & 31;
    const int ty = threadIdx.x >> 5;

#if __has_builtin(__builtin_amdgcn_tensor_load_to_lds)
    if (threadIdx.x < 32) {   // one wave issues the DMA (EXEC ignored by TDM)
        const unsigned long long ga =
            (unsigned long long)(const void*)(ys + (size_t)t0 * 4096 + q0);
        const unsigned int lds_off =
            (unsigned int)(unsigned long long)(const void*)&tile[0];
        v4u g0;
        g0[0] = 1u;                                     // count=1, user mode
        g0[1] = lds_off;                                // lds_addr (bytes)
        g0[2] = (unsigned int)ga;                       // global_addr[31:0]
        g0[3] = (unsigned int)((ga >> 32) & 0x01FFFFFFull) | 0x80000000u; // [56:32] | type=2
        v8i g1;
        g1[0] = 0x01120000;        // data_size=4B, pad_enable, pad_interval=32dw, pad_amount=1dw
        g1[1] = 0x10000000;        // tensor_dim0[15:0]=4096 in bits[31:16]
        g1[2] = 0x1F000000;        // tensor_dim1[15:0]=7936 in bits[31:16]
        g1[3] = 0x00200000;        // tile_dim0=32 in bits[31:16]
        g1[4] = 32;                // tile_dim1=32
        g1[5] = 4096;              // tensor_dim0_stride[31:0]
        g1[6] = 0;
        g1[7] = 0;
        v4i gz4 = {0, 0, 0, 0};    // 2D tensor: groups 2/3 unused
        v8i gz8 = {0, 0, 0, 0, 0, 0, 0, 0};
        __builtin_amdgcn_tensor_load_to_lds(g0, g1, gz4, gz4, gz8, 0);
        __builtin_amdgcn_s_wait_tensorcnt(0);
    }
    __syncthreads();
#else
    for (int yy = ty; yy < 32; yy += 8) {
        const int t = t0 + yy;
        if (t < NSTEPS) tile[yy * 33 + tx] = ys[(size_t)t * 4096 + q0 + tx];
    }
    __syncthreads();
#endif

    for (int yy = ty; yy < 32; yy += 8) {
        const int q   = q0 + yy;
        const int ot  = q >> 8;
        const int rem = q & 255;
        const int l   = rem >> 3;
        const int r   = rem & 7;
        const int o   = ot * 16 + r + ((l >> 4) << 3);
        const int b   = l & 15;
        const int t   = t0 + tx;
        if (t < NSTEPS) out[(size_t)(b * 256 + o) * NELEM + 1 + t] = tile[tx * 33 + yy];
    }
}

// ---------------------------------------------------------------------------
extern "C" void kernel_launch(void* const* d_in, const int* in_sizes, int n_in,
                              void* d_out, int out_size, void* d_ws, size_t ws_size,
                              hipStream_t stream) {
    (void)in_sizes; (void)n_in; (void)out_size; (void)ws_size;
    const float* x      = (const float*)d_in[0];
    const float* weight = (const float*)d_in[1];
    float*       out    = (float*)d_out;

    // Workspace layout (floats unless noted):
    float* cbuf = (float*)d_ws;                               // TPAD*4096
    char*  p    = (char*)d_ws + (size_t)TPAD * 4096 * 4;
    __bf16* wfrag = (__bf16*)p;            p += 2 * 65536 * 2;   // bf16 W frags
    float* w0f32  = (float*)p;             p += 65536 * 4;       // W0 f32 frags
    float* w0Lf32 = (float*)p;             p += 65536 * 4;       // W0^L f32 frags
    float* Pa     = (float*)p;             p += 65536 * 4;       // power ping
    float* Pb     = (float*)p;             p += 65536 * 4;       // power pong
    float* estates = (float*)p;            p += (size_t)(GCHUNKS - 1) * 4096 * 4;
    float* sstates = (float*)p;            p += (size_t)GCHUNKS * 4096 * 4;
    __bf16* w1f = wfrag + 65536;           // tap 1 (conv)

    (void)hipMemcpyAsync(out, x, (size_t)B_ * C_ * NELEM * sizeof(float),
                         hipMemcpyDeviceToDevice, stream);

    prep_wfrag_kernel<<<dim3(2, 16), 512, 0, stream>>>(weight, wfrag);
    extract_w0_kernel<<<256, 256, 0, stream>>>(weight, Pa);
    prep_f32frag_kernel<<<16, 256, 0, stream>>>(Pa, w0f32);

    conv_c_kernel<<<dim3(TTILES, 16), 512, 0, stream>>>(x, w1f, cbuf);

    // W0^LCHUNK by repeated squaring (7 = log2(128) launches, ping-pong)
    const float* src = Pa; float* dst = Pb;
    for (int s = 0; s < 7; ++s) {
        matsq_kernel<<<dim3(16, 16), 32, 0, stream>>>(src, dst);
        const float* t = src; src = dst; dst = (float*)t;
    }
    prep_f32frag_kernel<<<16, 256, 0, stream>>>(src, w0Lf32);

    scan_pass1_kernel<<<GCHUNKS - 1, 512, 0, stream>>>(x, w0f32, cbuf, estates);
    scan_pass2_kernel<<<1, 512, 0, stream>>>(w0Lf32, estates, sstates);
    scan_pass3_kernel<<<GCHUNKS, 512, 0, stream>>>(x, w0f32, sstates, cbuf);

    transpose_out_kernel<<<dim3(248, 128), 256, 0, stream>>>(cbuf, out);
}
